// FastAttention_2980707303460
// MI455X (gfx1250) — compile-verified, matrix operand order checked by reading, not run on hardware
//
#include <hip/hip_runtime.h>
#include <hip/hip_bf16.h>

// ---------------- types ----------------
typedef __attribute__((ext_vector_type(16))) __bf16 v16bf;
typedef __attribute__((ext_vector_type(8)))  __bf16 v8bf;
typedef __attribute__((ext_vector_type(4)))  __bf16 v4bf;
typedef __attribute__((ext_vector_type(8)))  float  v8f;

// ---------------- problem constants ----------------
#define BB 4
#define NN 4096
#define DD 1024
#define HH 16
#define EE 64
#define BH (BB*HH)          // 64
#define BN (BB*NN)          // 16384

// epilogue flags
#define F_RELU   1
#define F_TRANS  2
#define F_SCALE  4
#define F_F32OUT 8
#define F_HEADT 16

#define AROW 40   // LDS row stride in bf16 elems (80 B, conflict-free for b128)

// ---------------- f32 -> bf16 convert (4 elems/thread) ----------------
__global__ void k_cvt_bf16(const float* __restrict__ s, __bf16* __restrict__ d, long n4) {
    long i = (long)blockIdx.x * blockDim.x + threadIdx.x;
    if (i < n4) {
        float4 f = ((const float4*)s)[i];
        v4bf o;
        o.x = (__bf16)f.x; o.y = (__bf16)f.y; o.z = (__bf16)f.z; o.w = (__bf16)f.w;
        ((v4bf*)d)[i] = o;
    }
}

// orth_proj [H,E,E] f32 -> per-head transposed bf16 [H, d, e]
__global__ void k_orthT(const float* __restrict__ orth, __bf16* __restrict__ dst) {
    int t = blockIdx.x * blockDim.x + threadIdx.x;   // H*E*E = 65536 threads
    int h = t >> 12, rem = t & 4095;
    int e = rem >> 6, dcol = rem & 63;
    dst[h * 4096 + dcol * 64 + e] = (__bf16)orth[h * 4096 + e * 64 + dcol];
}

// ================= BIG-TILE GEMM: 128x128 block, 4 waves, 64x64 per wave =================
// C[M,Nn] = A[M,K] (row-major, lda) x B^T, Bm stored [Nn,K] row-major (ldb).
// Register-staged pipeline: global loads for K-tile i+1 issue during compute of tile i.
__global__ __launch_bounds__(128)
void k_gemm_wmma_big(const __bf16* __restrict__ A, long lda,
                     const __bf16* __restrict__ Bm, long ldb,
                     long M, long Nn, long K,
                     int flags, __bf16* __restrict__ bd, float* __restrict__ fd,
                     long ldc, const float* __restrict__ bias)
{
    __shared__ __align__(16) __bf16 As[128 * AROW];
    __shared__ __align__(16) __bf16 Bs[128 * AROW];

    const int tid   = threadIdx.x;
    const int lane  = tid & 31;
    const int wave  = tid >> 5;       // 4 waves, 2x2 over 128x128
    const int waveM = wave >> 1;
    const int waveN = wave & 1;
    const long m0 = (long)blockIdx.x * 128;
    const long n0 = (long)blockIdx.y * 128;

    const int lrow = lane & 15;
    const int lhi  = lane >> 4;
    const int koff = lhi * 8;

    v8f acc[4][4];
    v8f zz = {0.f,0.f,0.f,0.f,0.f,0.f,0.f,0.f};
#pragma unroll
    for (int i = 0; i < 4; ++i)
#pragma unroll
        for (int j = 0; j < 4; ++j) acc[i][j] = zz;

    // each of 128 threads owns one 32-elem (64 B) row of As and of Bs
    const __bf16* ga = A  + (m0 + tid) * lda;   // + k0
    const __bf16* gb = Bm + (n0 + tid) * ldb;

    // prologue: stage K-tile 0 in registers (explicit temps -> distinct VGPRs)
    v8bf ra[4], rb[4];
#pragma unroll
    for (int c = 0; c < 4; ++c) {
        ra[c] = *(const v8bf*)(ga + c * 8);
        rb[c] = *(const v8bf*)(gb + c * 8);
    }

    for (long k0 = 0; k0 < K; k0 += 32) {
        __syncthreads();                 // previous iteration's fragment reads done
#pragma unroll
        for (int c = 0; c < 4; ++c) {
            *(v8bf*)&As[tid * AROW + c * 8] = ra[c];
            *(v8bf*)&Bs[tid * AROW + c * 8] = rb[c];
        }
        const long kn = k0 + 32;
        if (kn < K) {                    // issue next tile's loads; waited at next ds_store
#pragma unroll
            for (int c = 0; c < 4; ++c) {
                ra[c] = *(const v8bf*)(ga + kn + c * 8);
                rb[c] = *(const v8bf*)(gb + kn + c * 8);
            }
        }
        __syncthreads();                 // LDS tile ready

        v16bf bfrag[4];
#pragma unroll
        for (int tn = 0; tn < 4; ++tn) {
            int br = waveN * 64 + tn * 16 + lrow;
            v8bf lo = *(const v8bf*)&Bs[br * AROW + koff];
            v8bf hi = *(const v8bf*)&Bs[br * AROW + koff + 16];
            bfrag[tn] = __builtin_shufflevector(lo, hi, 0,1,2,3,4,5,6,7,8,9,10,11,12,13,14,15);
        }
#pragma unroll
        for (int tm = 0; tm < 4; ++tm) {
            int ar = waveM * 64 + tm * 16 + lrow;
            v8bf lo = *(const v8bf*)&As[ar * AROW + koff];
            v8bf hi = *(const v8bf*)&As[ar * AROW + koff + 16];
            v16bf af = __builtin_shufflevector(lo, hi, 0,1,2,3,4,5,6,7,8,9,10,11,12,13,14,15);
#pragma unroll
            for (int tn = 0; tn < 4; ++tn)
                acc[tm][tn] = __builtin_amdgcn_wmma_f32_16x16x32_bf16(
                    false, af, false, bfrag[tn], (short)0, acc[tm][tn], false, false);
        }
    }

    // epilogue — lane holds rows row0..row0+7 (consecutive) at column col
#pragma unroll
    for (int tm = 0; tm < 4; ++tm)
#pragma unroll
        for (int tn = 0; tn < 4; ++tn) {
            long row0 = m0 + waveM * 64 + tm * 16 + lhi * 8;
            long col  = n0 + waveN * 64 + tn * 16 + lrow;
            if (flags & F_F32OUT) {
                float bv = bias ? bias[col] : 0.f;
#pragma unroll
                for (int r = 0; r < 8; ++r)
                    fd[(row0 + r) * ldc + col] = acc[tm][tn][r] + bv;
            } else if (flags & F_HEADT) {
                // v^T head-split: rows are consecutive n -> one 16 B packed store
                v8bf pk;
#pragma unroll
                for (int r = 0; r < 8; ++r) pk[r] = (__bf16)acc[tm][tn][r];
                long b = row0 >> 12, n = row0 & 4095, h = col >> 6, e = col & 63;
                *(v8bf*)&bd[((b * HH + h) * 64 + e) * (long)NN + n] = pk;
            } else {
#pragma unroll
                for (int r = 0; r < 8; ++r)
                    bd[(row0 + r) * ldc + col] = (__bf16)acc[tm][tn][r];
            }
        }
}

// ================= batched small GEMM: 64x64 block, 4 waves, 32x32 per wave =================
__global__ __launch_bounds__(128)
void k_gemm_wmma(const __bf16* __restrict__ A, long lda, long aDiv, long aMod,
                 const __bf16* __restrict__ Bm, long ldb, long bDiv, long bMod,
                 int zdiv, long M, long Nn, long K,
                 int flags, __bf16* __restrict__ bd, float* __restrict__ fd,
                 long ldc, long cDiv, long cMod,
                 const float* __restrict__ rowscale, long rsStride,
                 const float* __restrict__ bias)
{
    __shared__ __align__(16) __bf16 As[64 * AROW];
    __shared__ __align__(16) __bf16 Bs[64 * AROW];

    const int z  = blockIdx.z;
    const int zq = z / zdiv;
    const int zr = z % zdiv;
    A  += (long)zq * aDiv + (long)zr * aMod;
    Bm += (long)zq * bDiv + (long)zr * bMod;
    const long cOff = (long)zq * cDiv + (long)zr * cMod;

    const int tid   = threadIdx.x;
    const int lane  = tid & 31;
    const int wave  = tid >> 5;
    const int waveM = wave >> 1;
    const int waveN = wave & 1;
    const long m0 = (long)blockIdx.x * 64;
    const long n0 = (long)blockIdx.y * 64;

    const int crow  = tid >> 1;           // 0..63
    const int chalf = (tid & 1) * 16;     // elem offset 0 or 16

    const int lrow = lane & 15;
    const int lhi  = lane >> 4;
    const int koff = lhi * 8;

    v8f acc[2][2];
    v8f zz = {0.f,0.f,0.f,0.f,0.f,0.f,0.f,0.f};
    acc[0][0] = zz; acc[0][1] = zz; acc[1][0] = zz; acc[1][1] = zz;

    const __bf16* ga = A  + (m0 + crow) * lda + chalf;   // + k0
    const __bf16* gb = Bm + (n0 + crow) * ldb + chalf;

    v8bf ra[2], rb[2];
    ra[0] = *(const v8bf*)ga;       ra[1] = *(const v8bf*)(ga + 8);
    rb[0] = *(const v8bf*)gb;       rb[1] = *(const v8bf*)(gb + 8);

    for (long k0 = 0; k0 < K; k0 += 32) {
        __syncthreads();
        *(v8bf*)&As[crow * AROW + chalf]     = ra[0];
        *(v8bf*)&As[crow * AROW + chalf + 8] = ra[1];
        *(v8bf*)&Bs[crow * AROW + chalf]     = rb[0];
        *(v8bf*)&Bs[crow * AROW + chalf + 8] = rb[1];
        const long kn = k0 + 32;
        if (kn < K) {
            ra[0] = *(const v8bf*)(ga + kn);  ra[1] = *(const v8bf*)(ga + kn + 8);
            rb[0] = *(const v8bf*)(gb + kn);  rb[1] = *(const v8bf*)(gb + kn + 8);
        }
        __syncthreads();

        v16bf af[2], bfr[2];
#pragma unroll
        for (int t = 0; t < 2; ++t) {
            int ar = waveM * 32 + t * 16 + lrow;
            v8bf lo = *(const v8bf*)&As[ar * AROW + koff];
            v8bf hi = *(const v8bf*)&As[ar * AROW + koff + 16];
            af[t] = __builtin_shufflevector(lo, hi, 0,1,2,3,4,5,6,7,8,9,10,11,12,13,14,15);
            int br = waveN * 32 + t * 16 + lrow;
            lo = *(const v8bf*)&Bs[br * AROW + koff];
            hi = *(const v8bf*)&Bs[br * AROW + koff + 16];
            bfr[t] = __builtin_shufflevector(lo, hi, 0,1,2,3,4,5,6,7,8,9,10,11,12,13,14,15);
        }
#pragma unroll
        for (int tm = 0; tm < 2; ++tm)
#pragma unroll
            for (int tn = 0; tn < 2; ++tn)
                acc[tm][tn] = __builtin_amdgcn_wmma_f32_16x16x32_bf16(
                    false, af[tm], false, bfr[tn], (short)0, acc[tm][tn], false, false);
    }

#pragma unroll
    for (int tm = 0; tm < 2; ++tm)
#pragma unroll
        for (int tn = 0; tn < 2; ++tn) {
            long row0 = m0 + waveM * 32 + tm * 16 + lhi * 8;
            long col  = n0 + waveN * 32 + tn * 16 + lrow;
            float vv[8];
#pragma unroll
            for (int r = 0; r < 8; ++r) {
                float v = acc[tm][tn][r];
                if (flags & F_RELU)  v = v > 0.f ? v : 0.f;
                if (flags & F_SCALE) v *= rowscale[(long)z * rsStride + row0 + r];
                vv[r] = v;
            }
            if (flags & F_F32OUT) {
                float bv = bias ? bias[col] : 0.f;
#pragma unroll
                for (int r = 0; r < 8; ++r)
                    fd[cOff + (row0 + r) * ldc + col] = vv[r] + bv;
            } else if (flags & F_HEADT) {
                v8bf pk;
#pragma unroll
                for (int r = 0; r < 8; ++r) pk[r] = (__bf16)vv[r];
                long b = row0 >> 12, n = row0 & 4095, h = col >> 6, e = col & 63;
                *(v8bf*)&bd[((b * HH + h) * 64 + e) * (long)NN + n] = pk;
            } else if (flags & F_TRANS) {
                // rows consecutive in transposed dest -> one 16 B packed store
                v8bf pk;
#pragma unroll
                for (int r = 0; r < 8; ++r) pk[r] = (__bf16)vv[r];
                *(v8bf*)&bd[cOff + col * ldc + row0] = pk;
            } else {
#pragma unroll
                for (int r = 0; r < 8; ++r)
                    bd[cOff + (row0 + r) * ldc + col] = (__bf16)vv[r];
            }
        }
}

// ---------------- phi_k row-sum: phikT [BH][64][N] -> ksum [BH*64] ----------------
__global__ void k_rowsum(const __bf16* __restrict__ phikT, float* __restrict__ ksum) {
    __shared__ float red[256];
    long rowid = blockIdx.x;
    const __bf16* p = phikT + rowid * (long)NN;
    float s = 0.f;
    for (int i = threadIdx.x; i < NN; i += 256) s += (float)p[i];
    red[threadIdx.x] = s;
    __syncthreads();
    for (int o = 128; o > 0; o >>= 1) {
        if (threadIdx.x < o) red[threadIdx.x] += red[threadIdx.x + o];
        __syncthreads();
    }
    if (threadIdx.x == 0) ksum[rowid] = red[0];
}

// ---------------- denom -> inv ----------------
__global__ void k_inv(const __bf16* __restrict__ phiq, const float* __restrict__ ksum,
                      float* __restrict__ inv) {
    long t = (long)blockIdx.x * blockDim.x + threadIdx.x;   // BH*N threads
    long bh = t >> 12, n = t & 4095;
    const __bf16* pq = phiq + (bh * NN + n) * (long)EE;
    const float*  ks = ksum + bh * EE;
    float acc = 0.f;
#pragma unroll 8
    for (int d = 0; d < EE; ++d) acc += (float)pq[d] * ks[d];
    inv[t] = 1.0f / (0.125f * acc + 1e-8f);   // scale = 1/sqrt(64)
}

// ---------------- driver ----------------
extern "C" void kernel_launch(void* const* d_in, const int* in_sizes, int n_in,
                              void* d_out, int out_size, void* d_ws, size_t ws_size,
                              hipStream_t stream) {
    (void)in_sizes; (void)n_in; (void)out_size; (void)ws_size;
    const float* query = (const float*)d_in[0];
    const float* Wq    = (const float*)d_in[1];
    const float* Wk    = (const float*)d_in[2];
    const float* Wv    = (const float*)d_in[3];
    const float* orth  = (const float*)d_in[4];
    const float* Wo    = (const float*)d_in[5];
    const float* bo    = (const float*)d_in[6];
    float* out = (float*)d_out;

    const long S_X = (long)BN * DD;
    const long S_W = (long)DD * DD;
    const long S_O = (long)HH * EE * EE;

    char* p = (char*)d_ws;
    auto alloc = [&](size_t bytes) { char* r = p; p += (bytes + 255) & ~(size_t)255; return r; };
    __bf16* Xbf  = (__bf16*)alloc(S_X * 2);    // aliased to phiq after QKV
    __bf16* qbf  = (__bf16*)alloc(S_X * 2);    // aliased to phikT after phi_q
    __bf16* kbf  = (__bf16*)alloc(S_X * 2);    // aliased to x2 after phi_k
    __bf16* vT   = (__bf16*)alloc(S_X * 2);
    __bf16* Wqb  = (__bf16*)alloc(S_W * 2);
    __bf16* Wkb  = (__bf16*)alloc(S_W * 2);
    __bf16* Wvb  = (__bf16*)alloc(S_W * 2);
    __bf16* Wob  = (__bf16*)alloc(S_W * 2);
    __bf16* orT  = (__bf16*)alloc(S_O * 2);
    __bf16* kvT  = (__bf16*)alloc((long)BH * EE * EE * 2);
    float*  ksum = (float*)alloc((long)BH * EE * 4);
    float*  inv  = (float*)alloc((long)BH * NN * 4);
    __bf16* phiq  = Xbf;   // [bh][n][64]
    __bf16* phikT = qbf;   // [bh][64][N]
    __bf16* x2    = kbf;   // [b][n][1024] merged heads

    // 1. convert inputs to bf16
    k_cvt_bf16<<<dim3((S_X/4 + 255)/256), 256, 0, stream>>>(query, Xbf, S_X/4);
    k_cvt_bf16<<<dim3((S_W/4 + 255)/256), 256, 0, stream>>>(Wq, Wqb, S_W/4);
    k_cvt_bf16<<<dim3((S_W/4 + 255)/256), 256, 0, stream>>>(Wk, Wkb, S_W/4);
    k_cvt_bf16<<<dim3((S_W/4 + 255)/256), 256, 0, stream>>>(Wv, Wvb, S_W/4);
    k_cvt_bf16<<<dim3((S_W/4 + 255)/256), 256, 0, stream>>>(Wo, Wob, S_W/4);
    k_orthT<<<dim3(S_O/256), 256, 0, stream>>>(orth, orT);

    // 2. q,k,v projections with big-tile kernel: [16384,1024] x W^T
    dim3 gBig(BN/128, DD/128, 1);
    k_gemm_wmma_big<<<gBig, 128, 0, stream>>>(Xbf, DD, Wqb, DD, BN, DD, DD,
        0, qbf, nullptr, DD, nullptr);
    k_gemm_wmma_big<<<gBig, 128, 0, stream>>>(Xbf, DD, Wkb, DD, BN, DD, DD,
        0, kbf, nullptr, DD, nullptr);
    k_gemm_wmma_big<<<gBig, 128, 0, stream>>>(Xbf, DD, Wvb, DD, BN, DD, DD,
        F_HEADT, vT, nullptr, 0, nullptr);

    // 3. phi_q / phi_k (batched per (b,h)), phi_k stored transposed
    dim3 gPhi(NN/64, 1, BH);
    const long zS = (long)NN * EE;  // 262144 per-(b,h) phi slab
    k_gemm_wmma<<<gPhi, 128, 0, stream>>>(qbf, DD, (long)NN*DD, EE,
        orT, EE, 0, (long)EE*EE, HH, NN, EE, EE,
        F_RELU, phiq, nullptr, EE, HH*zS, zS, nullptr, 0, nullptr);
    k_gemm_wmma<<<gPhi, 128, 0, stream>>>(kbf, DD, (long)NN*DD, EE,
        orT, EE, 0, (long)EE*EE, HH, NN, EE, EE,
        F_RELU | F_TRANS, phikT, nullptr, NN, HH*zS, zS, nullptr, 0, nullptr);

    // 4. phi_k row-sum and inverse denominator
    k_rowsum<<<dim3(BH*EE), 256, 0, stream>>>(phikT, ksum);
    k_inv<<<dim3((BH*NN)/256), 256, 0, stream>>>(phiq, ksum, inv);

    // 5. kv = phi_k^T v  -> store kv^T
    dim3 gKV(1, 1, BH);
    k_gemm_wmma<<<gKV, 128, 0, stream>>>(phikT, NN, zS, 0,
        vT, NN, zS, 0, 1, EE, EE, NN,
        F_TRANS, kvT, nullptr, EE, (long)EE*EE, 0, nullptr, 0, nullptr);

    // 6. attn = (phi_q @ kv) * inv[n], merged-head store
    k_gemm_wmma<<<gPhi, 128, 0, stream>>>(phiq, EE, HH*zS, zS,
        kvT, EE, (long)HH*EE*EE, (long)EE*EE, HH, NN, EE, EE,
        F_SCALE, x2, nullptr, DD, (long)NN*DD, EE, inv, NN, nullptr);

    // 7. out = x2 @ Wo^T + bo (f32) with big-tile kernel
    k_gemm_wmma_big<<<gBig, 128, 0, stream>>>(x2, DD, Wob, DD, BN, DD, DD,
        F_F32OUT, nullptr, out, DD, bo);
}
